// GCNLayer_85375359910350
// MI455X (gfx1250) — matrix operand order repaired and checked by reading, not compiled, not run on hardware
//
#include <hip/hip_runtime.h>
#include <hip/hip_bf16.h>
#include <cstdint>

typedef __attribute__((ext_vector_type(16))) _Float16 v16h;
typedef __attribute__((ext_vector_type(8)))  _Float16 v8h;
typedef __attribute__((ext_vector_type(8)))  float    v8f;

#define DIM 128

// ---------------------------------------------------------------------------
// 1) init: zero agg (=d_out) + stats, deg = 1.0 (self loop)
// ---------------------------------------------------------------------------
__global__ void gcn_init(float* __restrict__ agg, float* __restrict__ stats,
                         float* __restrict__ deg, int total, int n) {
  int t = blockIdx.x * blockDim.x + threadIdx.x;
  if (t < total) agg[t] = 0.0f;
  if (t < 512) stats[t] = 0.0f;
  if (t < n) deg[t] = 1.0f;   // self loop contributes 1 to degree
}

// ---------------------------------------------------------------------------
// 2) degree count over target nodes (col = edge_index[1])
// ---------------------------------------------------------------------------
__global__ void gcn_deg(const long long* __restrict__ ei, float* __restrict__ deg,
                        int E) {
  int t = blockIdx.x * blockDim.x + threadIdx.x;
  if (t >= E) return;
  int col = (int)ei[(size_t)E + t];
  atomicAdd(&deg[col], 1.0f);
}

// 3) deg -> deg^{-1/2} in place (deg >= 1 always, self loop)
__global__ void gcn_dinv(float* __restrict__ deg, int n) {
  int t = blockIdx.x * blockDim.x + threadIdx.x;
  if (t < n) deg[t] = rsqrtf(deg[t]);
}

// ---------------------------------------------------------------------------
// 4) f32 -> f16 conversion (vectorized x4)
// ---------------------------------------------------------------------------
__global__ void gcn_cvt16(const float* __restrict__ in, _Float16* __restrict__ out,
                          int total4) {
  int t = blockIdx.x * blockDim.x + threadIdx.x;
  if (t >= total4) return;
  float4 v = ((const float4*)in)[t];
  union { _Float16 h[4]; unsigned long long u; } p;
  p.h[0] = (_Float16)v.x; p.h[1] = (_Float16)v.y;
  p.h[2] = (_Float16)v.z; p.h[3] = (_Float16)v.w;
  ((unsigned long long*)out)[t] = p.u;
}

// ---------------------------------------------------------------------------
// 5) WMMA GEMM: h[M,128] = xh[M,128] @ Wh[128,128]^T   (f16 in, f32 out)
//    W staged into LDS via async-copy; A preloaded (4 frags) per strip;
//    B fragments come from LDS (ds_load_b128). Grid-stride over M strips.
// ---------------------------------------------------------------------------
__global__ __launch_bounds__(256) void gcn_gemm(const _Float16* __restrict__ xh,
                                                const _Float16* __restrict__ Wh,
                                                float* __restrict__ h, int M) {
  __shared__ _Float16 Wlds[DIM * DIM];   // 32 KB

  // --- async copy W (32768 bytes) global -> LDS: 2048 chunks of 16B ---
  {
    unsigned ldsBase = (unsigned)(uintptr_t)(void*)Wlds;
    unsigned long long gBase = (unsigned long long)(uintptr_t)Wh;
    for (int i = threadIdx.x; i < 2048; i += 256) {
      unsigned ldsAddr = ldsBase + (unsigned)i * 16u;
      unsigned gOff = (unsigned)i * 16u;
      asm volatile("global_load_async_to_lds_b128 %0, %1, %2"
                   :
                   : "v"(ldsAddr), "v"(gOff), "s"(gBase)
                   : "memory");
    }
    asm volatile("s_wait_asynccnt 0x0" ::: "memory");
  }
  __syncthreads();

  const int lane = threadIdx.x & 31;
  const int halfSel = lane >> 4;         // 0: lanes 0-15, 1: lanes 16-31
  const int sub = lane & 15;

  const int waveId = (blockIdx.x * blockDim.x + threadIdx.x) >> 5;
  const int nWaves = (gridDim.x * blockDim.x) >> 5;
  const int nStrips = M >> 4;

  for (int strip = waveId; strip < nStrips; strip += nWaves) {
    const int m0 = strip * 16;
    const _Float16* arow = xh + (size_t)(m0 + sub) * DIM;

    // preload all 4 A fragments (one s_clause of 8 global_load_b128)
    union { v16h v; v8h p[2]; } A[4];
    #pragma unroll
    for (int kk = 0; kk < 4; ++kk) {
      const _Float16* ap = arow + kk * 32 + halfSel * 8;
      A[kk].p[0] = *(const v8h*)(ap);
      A[kk].p[1] = *(const v8h*)(ap + 16);
    }

    v8f acc[8] = {};
    #pragma unroll
    for (int kk = 0; kk < 4; ++kk) {
      const int kb = kk * 32;
      #pragma unroll
      for (int nt = 0; nt < 8; ++nt) {
        // B fragment: column n = nt*16+sub; K halfs [8h..8h+7] and [16+8h..16+8h+7] from LDS
        union { v16h v; v8h p[2]; } B;
        const _Float16* bp = &Wlds[(nt * 16 + sub) * DIM + kb + halfSel * 8];
        B.p[0] = *(const v8h*)(bp);
        B.p[1] = *(const v8h*)(bp + 16);
        acc[nt] = __builtin_amdgcn_wmma_f32_16x16x32_f16(
            false, A[kk].v, false, B.v, (short)0, acc[nt], false, false);
      }
    }

    // C/D layout: VGPR r -> (M = r + 8*halfSel, N = sub) within tile
    #pragma unroll
    for (int nt = 0; nt < 8; ++nt) {
      float* orow = h + (size_t)(m0 + halfSel * 8) * DIM + nt * 16 + sub;
      #pragma unroll
      for (int r = 0; r < 8; ++r) orow[(size_t)r * DIM] = acc[nt][r];
    }
  }
}

// ---------------------------------------------------------------------------
// 6) edge scatter: agg[col] += h[row] * dinv[row]*dinv[col]
//    one wave per edge; each lane owns 4 channels. h + agg are L2-resident.
// ---------------------------------------------------------------------------
__global__ void gcn_scatter(const long long* __restrict__ ei,
                            const float* __restrict__ h,
                            const float* __restrict__ dinv,
                            float* __restrict__ agg, int E) {
  unsigned t = blockIdx.x * blockDim.x + threadIdx.x;
  int e = t >> 5;
  int lane = t & 31;
  if (e >= E) return;
  int row = (int)ei[e];
  int col = (int)ei[(size_t)E + e];
  float nrm = dinv[row] * dinv[col];
  const float4 v = *(const float4*)(h + (size_t)row * DIM + lane * 4);
  float* dst = agg + (size_t)col * DIM + lane * 4;
  atomicAdd(dst + 0, v.x * nrm);
  atomicAdd(dst + 1, v.y * nrm);
  atomicAdd(dst + 2, v.z * nrm);
  atomicAdd(dst + 3, v.w * nrm);
}

// ---------------------------------------------------------------------------
// 7) pre-BN: agg += h*dinv^2 (self loop) + bias ; accumulate BN partial sums
// ---------------------------------------------------------------------------
__global__ __launch_bounds__(256) void gcn_prebn(float* __restrict__ agg,
                                                 const float* __restrict__ h,
                                                 const float* __restrict__ dinv,
                                                 const float* __restrict__ bias,
                                                 float* __restrict__ stats,
                                                 int n, int rowStride) {
  int c = threadIdx.x & (DIM - 1);
  int r0 = blockIdx.x * 2 + (threadIdx.x >> 7);
  float b = bias[c];
  float s = 0.0f, s2 = 0.0f;
  for (int r = r0; r < n; r += rowStride) {
    float d = dinv[r];
    size_t idx = (size_t)r * DIM + c;
    float v = agg[idx] + h[idx] * d * d + b;
    agg[idx] = v;
    s += v;
    s2 += v * v;
  }
  atomicAdd(&stats[c], s);
  atomicAdd(&stats[DIM + c], s2);
}

// 8) finalize mean / scale
__global__ void gcn_bnfinal(float* __restrict__ stats,
                            const float* __restrict__ gamma, int n) {
  int c = threadIdx.x;
  float invN = 1.0f / (float)n;
  float mean = stats[c] * invN;
  float var = stats[DIM + c] * invN - mean * mean;
  stats[2 * DIM + c] = mean;
  stats[3 * DIM + c] = rsqrtf(var + 1e-5f) * gamma[c];
}

// 9) normalize + ReLU, in place on d_out (vectorized x4)
__global__ void gcn_bnapply(float* __restrict__ out,
                            const float* __restrict__ stats,
                            const float* __restrict__ beta, int total4) {
  int t = blockIdx.x * blockDim.x + threadIdx.x;
  if (t >= total4) return;
  int c = (t & 31) * 4;
  float4 v = ((float4*)out)[t];
  const float* mean = stats + 2 * DIM;
  const float* scale = stats + 3 * DIM;
  v.x = fmaxf((v.x - mean[c + 0]) * scale[c + 0] + beta[c + 0], 0.0f);
  v.y = fmaxf((v.y - mean[c + 1]) * scale[c + 1] + beta[c + 1], 0.0f);
  v.z = fmaxf((v.z - mean[c + 2]) * scale[c + 2] + beta[c + 2], 0.0f);
  v.w = fmaxf((v.w - mean[c + 3]) * scale[c + 3] + beta[c + 3], 0.0f);
  ((float4*)out)[t] = v;
}

// ---------------------------------------------------------------------------
static inline size_t al256(size_t x) { return (x + 255) & ~(size_t)255; }

extern "C" void kernel_launch(void* const* d_in, const int* in_sizes, int n_in,
                              void* d_out, int out_size, void* d_ws, size_t ws_size,
                              hipStream_t stream) {
  const float*      x     = (const float*)d_in[0];
  const long long*  ei    = (const long long*)d_in[1];   // int64 [2, E]
  const float*      W     = (const float*)d_in[2];
  const float*      bias  = (const float*)d_in[3];
  const float*      gamma = (const float*)d_in[4];
  const float*      beta  = (const float*)d_in[5];

  const int n = in_sizes[0] / DIM;       // 100000
  const int E = in_sizes[1] / 2;         // 1600000
  const int total = n * DIM;             // 12.8M

  // workspace carve-out
  uint8_t* w = (uint8_t*)d_ws;
  size_t o = 0;
  float*    deg   = (float*)(w + o);    o += al256((size_t)n * 4);
  float*    stats = (float*)(w + o);    o += al256(512 * 4);
  _Float16* Wh    = (_Float16*)(w + o); o += al256((size_t)DIM * DIM * 2);
  _Float16* xh    = (_Float16*)(w + o); o += al256((size_t)total * 2);
  float*    hbuf  = (float*)(w + o);    o += al256((size_t)total * 4);
  if (o > ws_size) return;

  float* agg = (float*)d_out;            // aggregate straight into output buffer

  const int TB = 256;

  // 1) init zeros / deg=1
  gcn_init<<<(total + TB - 1) / TB, TB, 0, stream>>>(agg, stats, deg, total, n);
  // 2) degree histogram
  gcn_deg<<<(E + TB - 1) / TB, TB, 0, stream>>>(ei, deg, E);
  // 3) deg^{-1/2}
  gcn_dinv<<<(n + TB - 1) / TB, TB, 0, stream>>>(deg, n);
  // 4) f16 conversions
  gcn_cvt16<<<((DIM * DIM / 4) + TB - 1) / TB, TB, 0, stream>>>(W, Wh, DIM * DIM / 4);
  gcn_cvt16<<<((total / 4) + TB - 1) / TB, TB, 0, stream>>>(x, xh, total / 4);
  // 5) WMMA GEMM: h = x @ W^T  (persistent: 256 blocks x 8 waves, grid-stride)
  gcn_gemm<<<256, TB, 0, stream>>>(xh, Wh, hbuf, n);
  // 6) edge aggregation (one wave per edge)
  {
    long long threads = (long long)E * 32;
    gcn_scatter<<<(unsigned)((threads + TB - 1) / TB), TB, 0, stream>>>(ei, hbuf, deg, agg, E);
  }
  // 7) self loop + bias + BN partial sums
  {
    const int NB = 256;                 // rowStride = 512
    gcn_prebn<<<NB, TB, 0, stream>>>(agg, hbuf, deg, bias, stats, n, NB * 2);
  }
  // 8) BN finalize
  gcn_bnfinal<<<1, DIM, 0, stream>>>(stats, gamma, n);
  // 9) normalize + ReLU
  gcn_bnapply<<<((total / 4) + TB - 1) / TB, TB, 0, stream>>>(agg, stats, beta, total / 4);
}
